// Encoder_74071005987156
// MI455X (gfx1250) — matrix operand order repaired and checked by reading, not compiled, not run on hardware
//
#include <hip/hip_runtime.h>
#include <hip/hip_bf16.h>

// ---------------------------------------------------------------------------
// Transformer encoder layer for MI455X (gfx1250, wave32, WMMA).
// S=4096, D=512, H=8, E=64, FFN=2048. bf16 WMMA 16x16x32, f32 accumulate.
// All WMMA operands are K-contiguous (weights pre-transposed to [N][K],
// V transposed to [D][S]) so every fragment is two aligned b128 loads.
// ---------------------------------------------------------------------------

#define S_LEN  4096
#define DMODEL 512
#define NHEAD  8
#define HDIM   64
#define FFDIM  2048
#define LN_EPS 1e-5f

typedef __attribute__((ext_vector_type(16))) __bf16 v16bf;
typedef __attribute__((ext_vector_type(8)))  __bf16 v8bf;
typedef __attribute__((ext_vector_type(8)))  float  v8f;

// --- WMMA wrapper ----------------------------------------------------------
__device__ __forceinline__ v8f wmma_bf16(v16bf a, v16bf b, v8f c) {
  // 8 args: (neg_a, A, neg_b, B, c_mod, C, reuse_a, reuse_b)
  return __builtin_amdgcn_wmma_f32_16x16x32_bf16(false, a, false, b,
                                                 (short)0, c, false, false);
}

__device__ __forceinline__ v16bf concat8(v8bf lo, v8bf hi) {
  return __builtin_shufflevector(lo, hi, 0, 1, 2, 3, 4, 5, 6, 7, 8, 9, 10, 11,
                                 12, 13, 14, 15);
}

// --- Fragment loaders (ISA 7.12.2 VGPR layouts, wave32) --------------------
// A (16x32, row-major, lda elems): lane<16 -> row=lane, K {0..7,16..23}
//                                  lane>=16 -> row=lane-16, K {8..15,24..31}
// Two aligned 16B loads per lane.
__device__ __forceinline__ v16bf load_a_frag(const __bf16* A, int lda,
                                             int row0, int k0, int lane) {
  const __bf16* p = A + (size_t)(row0 + (lane & 15)) * lda + k0
                      + ((lane & 16) ? 8 : 0);
  v8bf lo = *(const v8bf*)p;
  v8bf hi = *(const v8bf*)(p + 16);
  return concat8(lo, hi);
}

// B fragment from K-contiguous storage Bt[N][K] (ldt elems per row):
// lane col = lane&15 -> Bt row col0+lane&15; lane<16 -> K {0..7,16..23},
// lane>=16 -> K {8..15,24..31} (matches A). Two b128 loads.
__device__ __forceinline__ v16bf load_bT_frag(const __bf16* Bt, int ldt,
                                              int k0, int col0, int lane) {
  const __bf16* p = Bt + (size_t)(col0 + (lane & 15)) * ldt + k0
                       + ((lane & 16) ? 8 : 0);
  v8bf lo = *(const v8bf*)p;
  v8bf hi = *(const v8bf*)(p + 16);
  return concat8(lo, hi);
}

// --- f32 -> bf16 cast ------------------------------------------------------
__global__ __launch_bounds__(256) void cast_kernel(const float* __restrict__ in,
                                                   __bf16* __restrict__ out,
                                                   int n) {
  int i = blockIdx.x * blockDim.x + threadIdx.x;
  if (i < n) out[i] = (__bf16)in[i];
}

// f32 [K][N] -> bf16 [N][K] transpose-cast, 32x32 LDS tiles, batched over z.
__global__ __launch_bounds__(256) void tcast_kernel(const float* __restrict__ in,
                                                    __bf16* __restrict__ out,
                                                    int K, int N) {
  __shared__ __bf16 tile[32][33];
  const float* inb = in + (size_t)blockIdx.z * K * N;
  __bf16*      ob  = out + (size_t)blockIdx.z * K * N;
  int tk0 = blockIdx.y * 32, tn0 = blockIdx.x * 32;
#pragma unroll
  for (int r = threadIdx.y; r < 32; r += 8)
    tile[r][threadIdx.x] = (__bf16)inb[(size_t)(tk0 + r) * N + tn0 + threadIdx.x];
  __syncthreads();
#pragma unroll
  for (int r = threadIdx.y; r < 32; r += 8)
    ob[(size_t)(tn0 + r) * K + tk0 + threadIdx.x] = tile[threadIdx.x][r];
}

__global__ void zero_kernel(float* p) {
  if (threadIdx.x < 4) p[threadIdx.x] = 0.0f;
}

// --- bf16 WMMA GEMM: C = A[MxK] * Bt[NxK]^T (+bias,+residual,ReLU) ---------
// Each wave computes a 16x64 output tile (1 A frag shared by 4 B frags).
template <bool RESID, bool RELU, bool OUTF, bool OUTB, bool OUTBT>
__global__ __launch_bounds__(128) void gemm_kernel(
    const __bf16* __restrict__ A, const __bf16* __restrict__ Bt,
    const float* __restrict__ bias, const float* __restrict__ residual,
    float* __restrict__ outf, __bf16* __restrict__ outb,
    __bf16* __restrict__ outbt, int M, int N, int K, int lda) {
  const int lane = threadIdx.x & 31;
  const int wave = threadIdx.x >> 5;
  const int gw   = blockIdx.x * (blockDim.x >> 5) + wave;
  const int tn   = N >> 6;
  const int m0   = (gw / tn) << 4;
  const int n0   = (gw % tn) << 6;
  if (m0 >= M) return;

  v8f acc0 = {}, acc1 = {}, acc2 = {}, acc3 = {};
  for (int k0 = 0; k0 < K; k0 += 32) {
    v16bf a = load_a_frag(A, lda, m0, k0, lane);
    if (k0 + 32 < K)  // global_prefetch_b8 for next A chunk
      __builtin_prefetch(A + (size_t)(m0 + (lane & 15)) * lda + k0 + 32, 0, 1);
    acc0 = wmma_bf16(a, load_bT_frag(Bt, K, k0, n0 +  0, lane), acc0);
    acc1 = wmma_bf16(a, load_bT_frag(Bt, K, k0, n0 + 16, lane), acc1);
    acc2 = wmma_bf16(a, load_bT_frag(Bt, K, k0, n0 + 32, lane), acc2);
    acc3 = wmma_bf16(a, load_bT_frag(Bt, K, k0, n0 + 48, lane), acc3);
  }

  const int cl = lane & 15;
  const int hi = lane >> 4;
#pragma unroll
  for (int t = 0; t < 4; ++t) {
    v8f acc = (t == 0) ? acc0 : (t == 1) ? acc1 : (t == 2) ? acc2 : acc3;
    int   col = n0 + t * 16 + cl;
    float bb  = bias[col];
#pragma unroll
    for (int j = 0; j < 8; ++j) {
      int    row = m0 + j + (hi << 3);
      size_t idx = (size_t)row * N + col;
      float  v   = acc[j] + bb;
      if (RESID) v += residual[idx];
      if (RELU) v = fmaxf(v, 0.0f);
      if (OUTF) outf[idx] = v;
      if (OUTB) outb[idx] = (__bf16)v;
      if (OUTBT) outbt[(size_t)col * M + row] = (__bf16)v;
    }
  }
}

// --- Flash attention: one wave per (head, 16-query tile) -------------------
// Q/K bf16 [S, D] head-major concat; Vt bf16 [D, S] (per-head row blocks).
__global__ __launch_bounds__(128) void attn_kernel(
    const __bf16* __restrict__ Q, const __bf16* __restrict__ K,
    const __bf16* __restrict__ Vt, __bf16* __restrict__ O) {
  __shared__ __bf16 pbuf[4][16 * 32];  // per-wave P staging (C-layout -> A-layout)
  const int lane = threadIdx.x & 31;
  const int wave = threadIdx.x >> 5;
  const int gw   = blockIdx.x * 4 + wave;  // H * S/16 = 2048 waves
  const int head = gw >> 8;                // / (S/16)
  const int m0   = (gw & 255) << 4;
  const float scale = 0.125f;              // 1/sqrt(64)

  const __bf16* Qh  = Q + head * HDIM;
  const __bf16* Kh  = K + head * HDIM;
  const __bf16* Vth = Vt + (size_t)head * HDIM * S_LEN;
  __bf16* pw = pbuf[wave];

  const int hi = lane >> 4;
  const int cl = lane & 15;

  // Q tile fragments held for whole kernel (E = 64 = 2 x K32)
  v16bf qa0 = load_a_frag(Qh, DMODEL, m0, 0, lane);
  v16bf qa1 = load_a_frag(Qh, DMODEL, m0, 32, lane);

  v8f o0 = {}, o1 = {}, o2 = {}, o3 = {};
  float mrow[8], lrow[8];
#pragma unroll
  for (int j = 0; j < 8; ++j) { mrow[j] = -__builtin_inff(); lrow[j] = 0.0f; }

  for (int kb = 0; kb < S_LEN; kb += 32) {
    // logits: B^T = K itself ([key][e] is K-contiguous). 2 tiles x 2 K-steps.
    v8f s0 = {}, s1 = {};
    s0 = wmma_bf16(qa0, load_bT_frag(Kh, DMODEL,  0, kb,      lane), s0);
    s0 = wmma_bf16(qa1, load_bT_frag(Kh, DMODEL, 32, kb,      lane), s0);
    s1 = wmma_bf16(qa0, load_bT_frag(Kh, DMODEL,  0, kb + 16, lane), s1);
    s1 = wmma_bf16(qa1, load_bT_frag(Kh, DMODEL, 32, kb + 16, lane), s1);

    // online softmax; C-layout: VGPR j -> row j+8*hi, col = cl.
    // 16-lane butterflies (xor 1,2,4,8) stay inside each wave32 half.
#pragma unroll
    for (int j = 0; j < 8; ++j) {
      float a = s0[j] * scale;
      float b = s1[j] * scale;
      float mx = fmaxf(a, b);
      mx = fmaxf(mx, __shfl_xor(mx, 1, 32));
      mx = fmaxf(mx, __shfl_xor(mx, 2, 32));
      mx = fmaxf(mx, __shfl_xor(mx, 4, 32));
      mx = fmaxf(mx, __shfl_xor(mx, 8, 32));
      float mnew = fmaxf(mrow[j], mx);
      float fac  = __expf(mrow[j] - mnew);
      mrow[j] = mnew;
      float p0 = __expf(a - mnew);
      float p1 = __expf(b - mnew);
      float rs = p0 + p1;
      rs += __shfl_xor(rs, 1, 32);
      rs += __shfl_xor(rs, 2, 32);
      rs += __shfl_xor(rs, 4, 32);
      rs += __shfl_xor(rs, 8, 32);
      lrow[j] = lrow[j] * fac + rs;
      o0[j] *= fac; o1[j] *= fac; o2[j] *= fac; o3[j] *= fac;
      int row = j + (hi << 3);
      pw[row * 32 + cl]      = (__bf16)p0;  // ds_store (per-wave region)
      pw[row * 32 + cl + 16] = (__bf16)p1;
    }

    // P[16x32] @ V[32x64]; V fragments from Vt are K(=key)-contiguous.
    v16bf pa = load_a_frag(pw, 32, 0, 0, lane);
    o0 = wmma_bf16(pa, load_bT_frag(Vth, S_LEN, kb,  0, lane), o0);
    o1 = wmma_bf16(pa, load_bT_frag(Vth, S_LEN, kb, 16, lane), o1);
    o2 = wmma_bf16(pa, load_bT_frag(Vth, S_LEN, kb, 32, lane), o2);
    o3 = wmma_bf16(pa, load_bT_frag(Vth, S_LEN, kb, 48, lane), o3);
  }

#pragma unroll
  for (int j = 0; j < 8; ++j) {
    float  inv = 1.0f / lrow[j];
    size_t row = (size_t)(m0 + j + (hi << 3)) * DMODEL + head * HDIM + cl;
    O[row +  0] = (__bf16)(o0[j] * inv);
    O[row + 16] = (__bf16)(o1[j] * inv);
    O[row + 32] = (__bf16)(o2[j] * inv);
    O[row + 48] = (__bf16)(o3[j] * inv);
  }
}

// --- Global LayerNorm over the full [S, D] tensor --------------------------
__global__ __launch_bounds__(256) void reduce_kernel(const float* __restrict__ z,
                                                     float* __restrict__ acc,
                                                     int n) {
  __shared__ float s0[256], s1[256];
  float a = 0.0f, b = 0.0f;
  for (int i = blockIdx.x * blockDim.x + threadIdx.x; i < n;
       i += gridDim.x * blockDim.x) {
    float v = z[i];
    a += v;
    b += v * v;
  }
  s0[threadIdx.x] = a;
  s1[threadIdx.x] = b;
  __syncthreads();
  for (int o = 128; o > 0; o >>= 1) {
    if (threadIdx.x < o) {
      s0[threadIdx.x] += s0[threadIdx.x + o];
      s1[threadIdx.x] += s1[threadIdx.x + o];
    }
    __syncthreads();
  }
  if (threadIdx.x == 0) {
    atomicAdd(&acc[0], s0[0]);
    atomicAdd(&acc[1], s1[0]);
  }
}

__global__ __launch_bounds__(256) void ln_apply_kernel(
    const float* __restrict__ z, const float* __restrict__ g,
    const float* __restrict__ b, const float* __restrict__ acc,
    float* __restrict__ outf, __bf16* __restrict__ outb, int n, float invN) {
  int i = blockIdx.x * blockDim.x + threadIdx.x;
  if (i >= n) return;
  float mu  = acc[0] * invN;
  float var = acc[1] * invN - mu * mu;
  float rs  = rsqrtf(var + LN_EPS);
  float v   = (z[i] - mu) * rs * g[i] + b[i];
  outf[i] = v;
  if (outb) outb[i] = (__bf16)v;
}

// ---------------------------------------------------------------------------
template <bool RESID, bool RELU, bool OUTF, bool OUTB, bool OUTBT>
static void launch_gemm(hipStream_t s, const __bf16* A, const __bf16* Bt,
                        const float* bias, const float* resid, float* outf,
                        __bf16* outb, __bf16* outbt, int M, int N, int K,
                        int lda) {
  int blocks = (M / 16) * (N / 64) / 4;  // 128 threads = 4 waves/block
  gemm_kernel<RESID, RELU, OUTF, OUTB, OUTBT><<<blocks, 128, 0, s>>>(
      A, Bt, bias, resid, outf, outb, outbt, M, N, K, lda);
}

extern "C" void kernel_launch(void* const* d_in, const int* in_sizes, int n_in,
                              void* d_out, int out_size, void* d_ws,
                              size_t ws_size, hipStream_t stream) {
  (void)in_sizes; (void)n_in; (void)out_size; (void)ws_size;
  const int Sn = S_LEN, D = DMODEL, E = HDIM, F = FFDIM;
  const int SD = Sn * D;

  const float* x    = (const float*)d_in[0];
  const float* Wq   = (const float*)d_in[1];
  const float* bq   = (const float*)d_in[2];
  const float* Wk   = (const float*)d_in[3];
  const float* bk   = (const float*)d_in[4];
  const float* Wv   = (const float*)d_in[5];
  const float* bv   = (const float*)d_in[6];
  const float* Wo   = (const float*)d_in[7];
  const float* bo   = (const float*)d_in[8];
  const float* W1   = (const float*)d_in[9];
  const float* b1   = (const float*)d_in[10];
  const float* W2   = (const float*)d_in[11];
  const float* b2   = (const float*)d_in[12];
  const float* ln_g = (const float*)d_in[13];
  const float* ln_b = (const float*)d_in[14];

  float* out   = (float*)d_out;  // [final | Kp | Vp], each S*D f32
  float* outKp = out + SD;
  float* outVp = out + 2 * SD;

  // workspace carve-out
  char*  ws  = (char*)d_ws;
  size_t off = 0;
  auto alloc = [&](size_t bytes) -> void* {
    void* p = ws + off;
    off += (bytes + 255) & ~(size_t)255;
    return p;
  };
  __bf16* xb    = (__bf16*)alloc((size_t)SD * 2);
  __bf16* Wqt   = (__bf16*)alloc((size_t)D * D * 2);  // [H*E][D]
  __bf16* Wkt   = (__bf16*)alloc((size_t)D * D * 2);
  __bf16* Wvt   = (__bf16*)alloc((size_t)D * D * 2);
  __bf16* Wot   = (__bf16*)alloc((size_t)D * D * 2);  // [D][D] (Wo^T)
  __bf16* W1t   = (__bf16*)alloc((size_t)D * F * 2);  // [F][D]
  __bf16* W2t   = (__bf16*)alloc((size_t)F * D * 2);  // [D][F]
  __bf16* Qb    = (__bf16*)alloc((size_t)SD * 2);     // [S][D] head-major
  __bf16* Kb    = (__bf16*)alloc((size_t)SD * 2);
  __bf16* Vb    = (__bf16*)alloc((size_t)SD * 2);
  __bf16* Vt    = (__bf16*)alloc((size_t)SD * 2);     // [D][S]
  __bf16* Ob    = (__bf16*)alloc((size_t)SD * 2);
  float*  res1  = (float*)alloc((size_t)SD * 4);
  float*  out1f = (float*)alloc((size_t)SD * 4);
  __bf16* out1b = (__bf16*)alloc((size_t)SD * 2);
  __bf16* h1b   = (__bf16*)alloc((size_t)Sn * F * 2);
  float*  res2  = (float*)alloc((size_t)SD * 4);
  float*  accb  = (float*)alloc(4 * sizeof(float));

  zero_kernel<<<1, 32, 0, stream>>>(accb);

  // casts / weight transposes ([K][N] f32 -> [N][K] bf16)
  cast_kernel<<<(SD + 255) / 256, 256, 0, stream>>>(x, xb, SD);
  dim3 tb(32, 8);
  tcast_kernel<<<dim3(E / 32, D / 32, NHEAD), tb, 0, stream>>>(Wq, Wqt, D, E);
  tcast_kernel<<<dim3(E / 32, D / 32, NHEAD), tb, 0, stream>>>(Wk, Wkt, D, E);
  tcast_kernel<<<dim3(E / 32, D / 32, NHEAD), tb, 0, stream>>>(Wv, Wvt, D, E);
  tcast_kernel<<<dim3(D / 32, D / 32, 1), tb, 0, stream>>>(Wo, Wot, D, D);
  tcast_kernel<<<dim3(F / 32, D / 32, 1), tb, 0, stream>>>(W1, W1t, D, F);
  tcast_kernel<<<dim3(D / 32, F / 32, 1), tb, 0, stream>>>(W2, W2t, F, D);

  // QKV projections -> head-major [S, H*E]; V also written transposed [D][S]
  launch_gemm<false, false, false, true, false>(stream, xb, Wqt, bq, nullptr,
                                                nullptr, Qb, nullptr, Sn, D, D, D);
  launch_gemm<false, false, false, true, false>(stream, xb, Wkt, bk, nullptr,
                                                nullptr, Kb, nullptr, Sn, D, D, D);
  launch_gemm<false, false, false, true, true>(stream, xb, Wvt, bv, nullptr,
                                               nullptr, Vb, Vt, Sn, D, D, D);

  // attention: 2048 waves, 4 per block
  attn_kernel<<<(NHEAD * Sn / 16) / 4, 128, 0, stream>>>(Qb, Kb, Vt, Ob);

  // res1 = x + attnOut @ Wo + bo ; Kp/Vp = {K,V}cat @ Wo + bo
  launch_gemm<true, false, true, false, false>(stream, Ob, Wot, bo, x, res1,
                                               nullptr, nullptr, Sn, D, D, D);
  launch_gemm<false, false, true, false, false>(stream, Kb, Wot, bo, nullptr,
                                                outKp, nullptr, nullptr, Sn, D, D, D);
  launch_gemm<false, false, true, false, false>(stream, Vb, Wot, bo, nullptr,
                                                outVp, nullptr, nullptr, Sn, D, D, D);

  // out1 = LayerNorm_{S,D}(res1)  (global mean/var over all S*D elements)
  reduce_kernel<<<1024, 256, 0, stream>>>(res1, accb, SD);
  ln_apply_kernel<<<(SD + 255) / 256, 256, 0, stream>>>(
      res1, ln_g, ln_b, accb, out1f, out1b, SD, 1.0f / (float)SD);

  // FFN: h1 = relu(out1 @ W1 + b1); res2 = out1 + h1 @ W2 + b2
  launch_gemm<false, true, false, true, false>(stream, out1b, W1t, b1, nullptr,
                                               nullptr, h1b, nullptr, Sn, F, D, D);
  launch_gemm<true, false, true, false, false>(stream, h1b, W2t, b2, out1f,
                                               res2, nullptr, nullptr, Sn, D, F, F);

  // final = LayerNorm_{S,D}(res2)
  reduce_kernel<<<1024, 256, 0, stream>>>(res2, accb + 2, SD);
  ln_apply_kernel<<<(SD + 255) / 256, 256, 0, stream>>>(
      res2, ln_g, ln_b, accb + 2, out, nullptr, SD, 1.0f / (float)SD);
}
